// TargetWindowEncoder_19808389169805
// MI455X (gfx1250) — matrix operand order repaired and checked by reading, not compiled, not run on hardware
//
#include <hip/hip_runtime.h>
#include <hip/hip_bf16.h>

// ---------------------------------------------------------------------------
// TargetWindowEncoder: 2-layer LSTM over 33-token sliding windows.
// S=512, B=8 -> N=4096 rows; E=H=512; gates G=2048; T=33 steps.
// bf16 WMMA (v_wmma_f32_16x16x32_bf16) with f32 accumulation.
// A-tiles staged block-wide into LDS via global_load_async_to_lds_b128.
// Workspace requirement: ~45 MiB.
// ---------------------------------------------------------------------------

typedef __attribute__((ext_vector_type(16))) __bf16 v16bf;
typedef __attribute__((ext_vector_type(8)))  __bf16 v8bf;
typedef __attribute__((ext_vector_type(8)))  float  v8f;
typedef __attribute__((ext_vector_type(4)))  int    v4i_;

#define NR  4096     // S*B rows
#define HD  512      // hidden == embed dim
#define GD  2048     // 4*HD gate dim
#define WW  32       // window
#define TT  33       // steps (W+1)
#define BB  8        // batch

#if defined(__has_builtin)
#  if __has_builtin(__builtin_amdgcn_global_load_async_to_lds_b128) && \
      __has_builtin(__builtin_amdgcn_s_wait_asynccnt)
#    define USE_ASYNC_LDS 1
#  endif
#endif
#ifndef USE_ASYNC_LDS
#  define USE_ASYNC_LDS 0
#endif

#if USE_ASYNC_LDS
typedef __attribute__((address_space(1))) v4i_* gv4p;   // global int4*
typedef __attribute__((address_space(3))) v4i_* lv4p;   // LDS int4*
#endif

// ---- helpers --------------------------------------------------------------

__device__ __forceinline__ unsigned short f2bf(float f) {
  unsigned int u = __float_as_uint(f);
  u += 0x7FFFu + ((u >> 16) & 1u);          // round-to-nearest-even
  return (unsigned short)(u >> 16);
}
__device__ __forceinline__ float sigf(float x) { return 1.0f / (1.0f + __expf(-x)); }
__device__ __forceinline__ float tanhfast(float x) {
  float e = __expf(2.0f * x);
  return 1.0f - 2.0f / (e + 1.0f);          // -> +/-1 at extremes, no NaN
}

// ---- setup kernels --------------------------------------------------------

__global__ void k_embed(const int* __restrict__ tok, const float* __restrict__ tbl,
                        unsigned short* __restrict__ out) {
  long i = (long)blockIdx.x * blockDim.x + threadIdx.x;   // exact grid: NR*HD
  int n = (int)(i >> 9);
  int e = (int)(i & 511);
  int t = tok[n];
  out[i] = f2bf(tbl[(long)t * HD + e]);
}

// Transpose (GD x HD) f32 -> (HD x GD) bf16 so WMMA B-fragments are contiguous.
__global__ void k_transpose(const float* __restrict__ w, unsigned short* __restrict__ wT) {
  long i = (long)blockIdx.x * blockDim.x + threadIdx.x;   // exact grid: HD*GD
  int k = (int)(i >> 11);
  int g = (int)(i & 2047);
  wT[i] = f2bf(w[(long)g * HD + k]);
}

__global__ void k_bias(const float* __restrict__ a, const float* __restrict__ b,
                       float* __restrict__ o) {
  int i = blockIdx.x * blockDim.x + threadIdx.x;
  if (i < GD) o[i] = a[i] + b[i];
}

__global__ void k_zero(float4* __restrict__ p, long n16) {
  long i = (long)blockIdx.x * blockDim.x + threadIdx.x;
  if (i < n16) p[i] = make_float4(0.f, 0.f, 0.f, 0.f);
}

// ---- A-tile staging: 64 rows x 32 K bf16 (4 KiB) per chunk ----------------
// 128 threads, 32 B each (two b128 transfers).
__device__ __forceinline__ void stage_a(unsigned short (* __restrict__ dst)[32],
                                        const unsigned short* __restrict__ abase,
                                        int rowOff, int M0, int tid, int ko) {
  int row  = tid >> 1;
  int half = (tid & 1) * 16;
  int gr = M0 + row + rowOff;                    // shifted-window row (layer 0)
  gr = gr < 0 ? 0 : (gr > NR - 1 ? NR - 1 : gr); // clamp; masked later anyway
  const unsigned short* src = abase + (long)gr * HD + ko + half;
  unsigned short* d = &dst[row][half];
#if USE_ASYNC_LDS
  __builtin_amdgcn_global_load_async_to_lds_b128(
      (gv4p)(unsigned short*)src, (lv4p)d, 0, 0);
  __builtin_amdgcn_global_load_async_to_lds_b128(
      (gv4p)((unsigned short*)src + 8), (lv4p)(d + 8), 0, 0);
#else
  uint4 v0 = *(const uint4*)src;
  uint4 v1 = *(const uint4*)(src + 8);
  *(uint4*)d = v0;
  *(uint4*)(d + 8) = v1;
#endif
}

// ---- GEMM phase: acc[2][4] (64x32 tile) += A(64xHD) * B(HDx32) ------------
// A frag (16-bit 16x32, ISA 7.12.2): lane l: M=l&15, dw0-3 = K kb..kb+7,
//   dw4-7 = K kb+16..kb+23, kb = 8*(l>>4) -- read from LDS (ds_load_b128).
// B frag (32x16): lane l holds N=l&15; K order same as A frag (strided gather).
__device__ __forceinline__ void gemm_phase(
    const unsigned short* __restrict__ abase, int rowOff,
    const unsigned short* __restrict__ bbase, int gcol0,
    int M0, int lm, int lh, int tid,
    unsigned short (* __restrict__ a_lds)[64][32],   // [2][64][32]
    v8f acc[2][4])
{
  const int kb = lh * 8;
  const __bf16* bp = (const __bf16*)bbase + gcol0 + lm;

  stage_a(a_lds[0], abase, rowOff, M0, tid, 0);
  for (int kc = 0; kc < 16; ++kc) {
    const int ko = kc * 32;
#if USE_ASYNC_LDS
    __builtin_amdgcn_s_wait_asynccnt(0);     // own stage done
#endif
    __syncthreads();                         // all waves staged + prior reads done
    if (kc + 1 < 16)
      stage_a(a_lds[(kc + 1) & 1], abase, rowOff, M0, tid, ko + 32);

    // B fragments: two 16-wide column tiles, column lm per lane, K gathered.
    const __bf16* bpk = bp + (long)ko * GD;
    v16bf b0, b1;
#pragma unroll
    for (int s = 0; s < 16; ++s) {
      const int k = (s < 8) ? (kb + s) : (8 + kb + s);
      b0[s] = bpk[(long)k * GD];
      b1[s] = bpk[(long)k * GD + 16];
    }
    if (kc + 1 < 16)
      __builtin_prefetch(bpk + (long)32 * GD, 0, 3);

    unsigned short (* __restrict__ ab)[32] = a_lds[kc & 1];
#pragma unroll
    for (int a = 0; a < 4; ++a) {
      const unsigned short* ap = &ab[a * 16 + lm][kb];
      v16bf af;
      ((v8bf*)&af)[0] = *(const v8bf*)(ap);        // ds_load_b128
      ((v8bf*)&af)[1] = *(const v8bf*)(ap + 16);   // ds_load_b128
      acc[0][a] = __builtin_amdgcn_wmma_f32_16x16x32_bf16(
          false, af, false, b0, (short)0, acc[0][a], false, false);
      acc[1][a] = __builtin_amdgcn_wmma_f32_16x16x32_bf16(
          false, af, false, b1, (short)0, acc[1][a], false, false);
    }
  }
  __syncthreads();   // protect buf0 before next phase restages it
}

// ---- fused LSTM step ------------------------------------------------------
// Block = 128 thr = 4 waves; wave g computes gate g's 64x32 tile.
// Grid = (NR/64, HD/32) = (64, 16).
__global__ __launch_bounds__(128)
void k_lstm_step(const unsigned short* __restrict__ xsrc, int xRowOff, int validRowStart,
                 const unsigned short* __restrict__ wTi, const unsigned short* __restrict__ wTh,
                 const float* __restrict__ bias,
                 const unsigned short* __restrict__ hprev, unsigned short* __restrict__ hnext,
                 float* __restrict__ c, float* __restrict__ hf32)
{
  __shared__ unsigned short a_lds[2][64][32];   // 8 KiB staging, double-buffered
  __shared__ float glds[4][64][32];             // 32 KiB gate exchange
  const int tid  = threadIdx.x;
  const int lane = tid & 31;
  const int gi   = tid >> 5;             // gate: 0=i 1=f 2=g 3=o
  const int lm   = lane & 15;
  const int lh   = lane >> 4;
  const int M0   = blockIdx.x * 64;
  const int j0   = blockIdx.y * 32;
  const int gcol0 = gi * HD + j0;

  v8f acc[2][4] = {};
  gemm_phase(xsrc,  xRowOff, wTi, gcol0, M0, lm, lh, tid, a_lds, acc);  // input
  gemm_phase(hprev, 0,       wTh, gcol0, M0, lm, lh, tid, a_lds, acc);  // recurrent

  // C/D layout: vgpr r -> row r + 8*lh (within 16-row subtile), col lm.
  float bv0 = bias[gcol0 + lm];
  float bv1 = bias[gcol0 + 16 + lm];
#pragma unroll
  for (int a = 0; a < 4; ++a)
#pragma unroll
    for (int r = 0; r < 8; ++r) {
      glds[gi][a * 16 + r + 8 * lh][lm]      = acc[0][a][r] + bv0;
      glds[gi][a * 16 + r + 8 * lh][16 + lm] = acc[1][a][r] + bv1;
    }
  __syncthreads();

  for (int idx = tid; idx < 64 * 32; idx += 128) {
    int row = idx >> 5, col = idx & 31;
    int n = M0 + row, j = j0 + col;
    float iv = sigf(glds[0][row][col]);
    float fv = sigf(glds[1][row][col]);
    float gv = tanhfast(glds[2][row][col]);
    float ov = sigf(glds[3][row][col]);
    long off = (long)n * HD + j;
    float cn = fv * c[off] + iv * gv;
    float hn = ov * tanhfast(cn);
    if (n < validRowStart) { cn = 0.f; hn = 0.f; }   // masked rows stay zero
    c[off] = cn;
    hnext[off] = f2bf(hn);
    if (hf32) hf32[off] = hn;                        // layer-1: stream to d_out
  }
}

// ---------------------------------------------------------------------------

extern "C" void kernel_launch(void* const* d_in, const int* in_sizes, int n_in,
                              void* d_out, int out_size, void* d_ws, size_t ws_size,
                              hipStream_t stream) {
  (void)in_sizes; (void)n_in; (void)out_size; (void)ws_size;
  const int*   tok  = (const int*)  d_in[0];
  const float* tbl  = (const float*)d_in[1];
  const float* wih0 = (const float*)d_in[2];
  const float* whh0 = (const float*)d_in[3];
  const float* bih0 = (const float*)d_in[4];
  const float* bhh0 = (const float*)d_in[5];
  const float* wih1 = (const float*)d_in[6];
  const float* whh1 = (const float*)d_in[7];
  const float* bih1 = (const float*)d_in[8];
  const float* bhh1 = (const float*)d_in[9];

  char* ws = (char*)d_ws;
  auto take = [&](size_t bytes) {
    char* p = ws; ws += (bytes + 255) & ~(size_t)255; return p;
  };
  unsigned short* xbf  = (unsigned short*)take((size_t)NR * HD * 2);  // emb bf16
  unsigned short* wT0i = (unsigned short*)take((size_t)HD * GD * 2);
  unsigned short* wT0h = (unsigned short*)take((size_t)HD * GD * 2);
  unsigned short* wT1i = (unsigned short*)take((size_t)HD * GD * 2);
  unsigned short* wT1h = (unsigned short*)take((size_t)HD * GD * 2);
  float* bias0 = (float*)take((size_t)GD * 4);
  float* bias1 = (float*)take((size_t)GD * 4);
  unsigned short* h0b[2] = {(unsigned short*)take((size_t)NR * HD * 2),
                            (unsigned short*)take((size_t)NR * HD * 2)};
  unsigned short* h1b[2] = {(unsigned short*)take((size_t)NR * HD * 2),
                            (unsigned short*)take((size_t)NR * HD * 2)};
  float* c0 = (float*)take((size_t)NR * HD * 4);
  float* c1 = (float*)take((size_t)NR * HD * 4);

  // --- setup ---
  k_embed    <<<(NR * HD) / 256, 256, 0, stream>>>(tok, tbl, xbf);
  k_transpose<<<(HD * GD) / 256, 256, 0, stream>>>(wih0, wT0i);
  k_transpose<<<(HD * GD) / 256, 256, 0, stream>>>(whh0, wT0h);
  k_transpose<<<(HD * GD) / 256, 256, 0, stream>>>(wih1, wT1i);
  k_transpose<<<(HD * GD) / 256, 256, 0, stream>>>(whh1, wT1h);
  k_bias<<<8, 256, 0, stream>>>(bih0, bhh0, bias0);
  k_bias<<<8, 256, 0, stream>>>(bih1, bhh1, bias1);
  long zh = (long)NR * HD * 2 / 16;   // bf16 state buffers, in float4 units
  long zc = (long)NR * HD * 4 / 16;   // f32 cell buffers
  k_zero<<<(unsigned)((zh + 255) / 256), 256, 0, stream>>>((float4*)h0b[0], zh);
  k_zero<<<(unsigned)((zh + 255) / 256), 256, 0, stream>>>((float4*)h1b[0], zh);
  k_zero<<<(unsigned)((zc + 255) / 256), 256, 0, stream>>>((float4*)c0, zc);
  k_zero<<<(unsigned)((zc + 255) / 256), 256, 0, stream>>>((float4*)c1, zc);

  // --- 33-step scan, 2 layers, h double-buffered ---
  dim3 grid(NR / 64, HD / 32);   // (64, 16)
  for (int t = 0; t < TT; ++t) {
    int vrs  = (WW - t) * BB;  if (vrs < 0) vrs = 0;  // first valid row
    int xoff = (t - WW) * BB;                         // shifted-window row offset
    int rp = t & 1, wp = (t + 1) & 1;
    k_lstm_step<<<grid, 128, 0, stream>>>(xbf, xoff, vrs, wT0i, wT0h, bias0,
                                          h0b[rp], h0b[wp], c0, nullptr);
    k_lstm_step<<<grid, 128, 0, stream>>>(h0b[wp], 0, vrs, wT1i, wT1h, bias1,
                                          h1b[rp], h1b[wp], c1, (float*)d_out);
  }
  // final step writes h1 (f32) into d_out -> (S, B, H) answer.
}
